// GTDamagePredictor_88880053224023
// MI455X (gfx1250) — compile-verified, matrix operand order checked by reading, not run on hardware
//
#include <hip/hip_runtime.h>
#include <hip/hip_bf16.h>
#include <math.h>

// ---------------------------------------------------------------------------
// GTDamagePredictor on MI455X (gfx1250): bf16 WMMA pipeline, v4.
// Fragments = 2x b128 K-contiguous loads; batched per-tile loads for dscnt
// overlap; branchless epilogues; async global->LDS staging (ASYNCcnt path).
// ---------------------------------------------------------------------------

typedef __attribute__((ext_vector_type(16))) __bf16 v16bf;
typedef __attribute__((ext_vector_type(8)))  __bf16 v8bf;
typedef __attribute__((ext_vector_type(8)))  float  v8f;

#define ALPHA 0.2f
#define NEGV  -9000000000000000.0f

#if defined(__has_builtin)
#  if __has_builtin(__builtin_amdgcn_global_load_async_to_lds_b64) && \
      __has_builtin(__builtin_amdgcn_s_wait_asynccnt)
#    define HAVE_ASYNC_LDS 1
#  endif
#endif
#ifndef HAVE_ASYNC_LDS
#  define HAVE_ASYNC_LDS 0
#endif

#if HAVE_ASYNC_LDS
// The builtin's pointee type encodes the transfer width: b64 -> 2 x int vector.
// Global source lives in AS1, LDS destination in AS3.
typedef int ai2 __attribute__((vector_size(8)));
typedef __attribute__((address_space(1))) ai2 g_ai2;
typedef __attribute__((address_space(3))) ai2 l_ai2;
#endif

__device__ __forceinline__ unsigned short f2bfu(float f) {
  unsigned int u = __builtin_bit_cast(unsigned int, f);
  unsigned int r = (u + 0x7FFFu + ((u >> 16) & 1u)) >> 16;   // RNE
  return (unsigned short)r;
}

// Fragment fast path (ISA 7.12.2): for lane-half h, element e holds
// K = kbase + h*8 + {0..7} (e<8) and K = kbase + 16 + h*8 + {0..7} (e>=8).
// With a K-contiguous source that is two aligned 16-byte loads.
__device__ __forceinline__ v16bf frag_ld(const unsigned short* p) {
  v8bf lo = *(const v8bf*)(p);
  v8bf hi = *(const v8bf*)(p + 16);
  return __builtin_shufflevector(lo, hi, 0, 1, 2, 3, 4, 5, 6, 7,
                                 8, 9, 10, 11, 12, 13, 14, 15);
}

__device__ __forceinline__ v8f wmma_bf16(v16bf a, v16bf b, v8f c) {
  return __builtin_amdgcn_wmma_f32_16x16x32_bf16(false, a, false, b, (short)0, c,
                                                 false, false);
}

// ---------------------------------------------------------------------------
// Kernel 1: conv1(2->32,k5,p2)+ReLU -> conv2(32->64,k5,p2)+ReLU -> mean(t)
// One block per (b,n) sample. conv2 as WMMA GEMM: M=64, K=160 (tap*32+ci), N=200.
// ---------------------------------------------------------------------------
__global__ __launch_bounds__(128)
void conv_fused_kernel(const float* __restrict__ xg,
                       const float* __restrict__ w1g, const float* __restrict__ b1g,
                       const float* __restrict__ w2g, const float* __restrict__ b2g,
                       float* __restrict__ hbar) {
  __shared__ __attribute__((aligned(16))) float          xs[204 * 2];   // [t][c]
  __shared__ __attribute__((aligned(16))) unsigned short h1t[216 * 32]; // [t][ci]
  __shared__ __attribute__((aligned(16))) unsigned short w2l[64 * 160]; // [co][tap*32+ci]
  __shared__ float w1l[320];
  __shared__ float b1l[32];
  __shared__ float b2l[64];
  __shared__ float accl[64];

  const int tid = threadIdx.x;
  const int s = blockIdx.x;
  const int b = s >> 9, n = s & 511;

  for (int i = tid; i < 216 * 32; i += 128) h1t[i] = 0;
  if (tid < 64) b2l[tid] = b2g[tid];
  if (tid < 32) b1l[tid] = b1g[tid];
  for (int i = tid; i < 320; i += 128) w1l[i] = w1g[i];
  for (int i = tid; i < 64 * 160; i += 128) {
    int co = i / 160, r = i - co * 160, ci = r / 5, kt = r - ci * 5;
    w2l[co * 160 + kt * 32 + ci] = f2bfu(w2g[i]);
  }
  if (tid < 4) { xs[tid] = 0.f; xs[404 + tid] = 0.f; }   // halo only
#if HAVE_ASYNC_LDS
  // Each lane copies one timestep's 2 contiguous floats straight to LDS.
  for (int t = tid; t < 200; t += 128) {
    __builtin_amdgcn_global_load_async_to_lds_b64(
        (g_ai2*)(xg + ((size_t)(b * 200 + t)) * 1024 + n * 2),
        (l_ai2*)&xs[(t + 2) * 2], 0, 0);
  }
  __builtin_amdgcn_s_wait_asynccnt(0);
#else
  for (int i = tid; i < 400; i += 128) {
    int t = i >> 1, c = i & 1;
    xs[(t + 2) * 2 + c] = xg[((size_t)(b * 200 + t)) * 1024 + n * 2 + c];
  }
#endif
  __syncthreads();

  // conv1 + ReLU (VALU; ~0.5 GFLOP total), stored transposed [t][ci]
  for (int i = tid; i < 32 * 200; i += 128) {
    int ci = i / 200, t = i - ci * 200;
    float acc = b1l[ci];
#pragma unroll
    for (int c = 0; c < 2; ++c)
#pragma unroll
      for (int k = 0; k < 5; ++k)
        acc += w1l[ci * 10 + c * 5 + k] * xs[(t + k) * 2 + c];
    h1t[(t + 2) * 32 + ci] = f2bfu(fmaxf(acc, 0.f));
  }
  __syncthreads();

  // conv2 via WMMA: wave w owns output rows co = 16w..16w+15
  const int wave = tid >> 5, lane = tid & 31;
  const int half = lane >> 4, ln = lane & 15;
  const unsigned short* abase = w2l + (wave * 16 + ln) * 160 + half * 8;

  v16bf afr[5];
#pragma unroll
  for (int kt = 0; kt < 5; ++kt) afr[kt] = frag_ld(abase + kt * 32);
  float bias[8];
#pragma unroll
  for (int r = 0; r < 8; ++r) bias[r] = b2l[wave * 16 + half * 8 + r];
  float sums[8];
#pragma unroll
  for (int r = 0; r < 8; ++r) sums[r] = 0.f;

  for (int nt = 0; nt < 13; ++nt) {            // 13 N-tiles cover t=0..207
    const unsigned short* bcol = h1t + (nt * 16 + ln) * 32 + half * 8;
    v16bf bfr[5];
#pragma unroll
    for (int kt = 0; kt < 5; ++kt) bfr[kt] = frag_ld(bcol + kt * 32); // batch loads
    v8f c = {};
#pragma unroll
    for (int kt = 0; kt < 5; ++kt) c = wmma_bf16(afr[kt], bfr[kt], c);
    float msk = (nt * 16 + ln < 200) ? 1.f : 0.f;  // branchless column mask
#pragma unroll
    for (int r = 0; r < 8; ++r) sums[r] += msk * fmaxf(c[r] + bias[r], 0.f);
  }
  // reduce over the 16 columns held by each lane-half (stays within half)
#pragma unroll
  for (int m = 1; m < 16; m <<= 1)
#pragma unroll
    for (int r = 0; r < 8; ++r) sums[r] += __shfl_xor(sums[r], m, 32);
  if (ln == 0) {
#pragma unroll
    for (int r = 0; r < 8; ++r) accl[wave * 16 + half * 8 + r] = sums[r];
  }
  __syncthreads();
  if (tid < 64) hbar[(size_t)s * 64 + tid] = accl[tid] * (1.f / 200.f);
}

// ---------------------------------------------------------------------------
// Kernel 2: Wh = h @ W. Also emits bf16 transposed copy Wht[b][o][j] for the
// attention GEMM's B fragments. Per block: (batch, 16-row tile); wave = N-tile.
// ---------------------------------------------------------------------------
__global__ __launch_bounds__(128)
void gat_wh_kernel(const float* __restrict__ hin, const float* __restrict__ Wg,
                   float* __restrict__ Wh, unsigned short* __restrict__ Wht) {
  __shared__ __attribute__((aligned(16))) unsigned short a_l[16 * 64];  // [m][k]
  __shared__ __attribute__((aligned(16))) unsigned short wl_t[64 * 64]; // [o][k]
  const int tid = threadIdx.x;
  const int b = blockIdx.x >> 5, mt = blockIdx.x & 31;
  const size_t rowbase = ((size_t)b * 512 + mt * 16);
  for (int i = tid; i < 16 * 64; i += 128) {
    int m = i >> 6, k = i & 63;
    a_l[i] = f2bfu(hin[(rowbase + m) * 64 + k]);
  }
  for (int i = tid; i < 64 * 64; i += 128) {
    int o = i >> 6, k = i & 63;
    wl_t[i] = f2bfu(Wg[k * 64 + o]);           // stage transposed
  }
  __syncthreads();
  const int wave = tid >> 5, lane = tid & 31;
  const int half = lane >> 4, ln = lane & 15;
  const int o = wave * 16 + ln;
  v16bf a0 = frag_ld(a_l + ln * 64 + half * 8);
  v16bf a1 = frag_ld(a_l + ln * 64 + 32 + half * 8);
  v16bf b0 = frag_ld(wl_t + o * 64 + half * 8);
  v16bf b1 = frag_ld(wl_t + o * 64 + 32 + half * 8);
  v8f c = {};
  c = wmma_bf16(a0, b0, c);
  c = wmma_bf16(a1, b1, c);
#pragma unroll
  for (int r = 0; r < 8; ++r) {
    int m = r + half * 8;
    size_t irow = rowbase + m;
    Wh[irow * 64 + o] = c[r];
    Wht[((size_t)b * 64 + o) * 512 + (mt * 16 + m)] = f2bfu(c[r]);
  }
}

// ---------------------------------------------------------------------------
// Kernel 3: f1 = Wh @ a[:H], f2 = Wh @ a[H:]
// ---------------------------------------------------------------------------
__global__ void gat_f_kernel(const float* __restrict__ Wh, const float* __restrict__ a,
                             float* __restrict__ f1, float* __restrict__ f2) {
  int idx = blockIdx.x * blockDim.x + threadIdx.x;   // 0..4095 = b*512+i
  const float* row = Wh + (size_t)idx * 64;
  float s1 = 0.f, s2 = 0.f;
#pragma unroll
  for (int o = 0; o < 64; ++o) { s1 += row[o] * a[o]; s2 += row[o] * a[64 + o]; }
  f1[idx] = s1; f2[idx] = s2;
}

// ---------------------------------------------------------------------------
// Kernel 4: e = leaky(f1_i + f2_j), mask, softmax_j -> att (bf16, LDS),
//           out = att @ Wh via K=512 WMMA chain (chunks of 4);
//           B frags = 2x global_load_b128 from L2-resident transposed Wht.
// ---------------------------------------------------------------------------
__global__ __launch_bounds__(128)
void gat_attn_kernel(const float* __restrict__ f1, const float* __restrict__ f2,
                     const int* __restrict__ adj,
                     const unsigned short* __restrict__ Wht,
                     float* __restrict__ out, int do_relu) {
  __shared__ float e_l[16 * 512];                                         // 32 KB
  __shared__ __attribute__((aligned(16))) unsigned short att_l[16 * 512]; // 16 KB
  const int tid = threadIdx.x;
  const int b = blockIdx.x >> 5, mt = blockIdx.x & 31;

  __builtin_prefetch(Wht + (size_t)b * 64 * 512 + tid * 256, 0, 1);

  for (int i = tid; i < 16 * 512; i += 128) {
    int il = i >> 9, j = i & 511;
    int ig = mt * 16 + il;
    float ev = f1[b * 512 + ig] + f2[b * 512 + j];
    ev = (ev > 0.f) ? ev : ALPHA * ev;
    if (adj[ig * 512 + j] == 0) ev = NEGV;
    e_l[i] = ev;
  }
  __syncthreads();
  {
    // softmax: 8 threads per row (contiguous tids -> same wave, shfl legal)
    int row = tid >> 3, sub = tid & 7;
    const float* er = e_l + row * 512;
    float mx = -3.4e38f;
    for (int j = sub; j < 512; j += 8) mx = fmaxf(mx, er[j]);
#pragma unroll
    for (int m = 1; m < 8; m <<= 1) mx = fmaxf(mx, __shfl_xor(mx, m, 32));
    float sum = 0.f;
    for (int j = sub; j < 512; j += 8) sum += expf(er[j] - mx);
#pragma unroll
    for (int m = 1; m < 8; m <<= 1) sum += __shfl_xor(sum, m, 32);
    float inv = 1.f / sum;
    unsigned short* ar = att_l + row * 512;
    for (int j = sub; j < 512; j += 8) ar[j] = f2bfu(expf(er[j] - mx) * inv);
  }
  __syncthreads();

  const int wave = tid >> 5, lane = tid & 31;
  const int half = lane >> 4, ln = lane & 15;
  const int o = wave * 16 + ln;
  const unsigned short* bt = Wht + ((size_t)b * 64 + o) * 512 + half * 8;
  const unsigned short* at = att_l + ln * 512 + half * 8;
  v8f c = {};
#pragma unroll
  for (int kb = 0; kb < 16; kb += 4) {         // pipeline chunks of 4 K-steps
    v16bf af[4], bf[4];
#pragma unroll
    for (int j = 0; j < 4; ++j) {
      af[j] = frag_ld(at + (kb + j) * 32);
      bf[j] = frag_ld(bt + (kb + j) * 32);
    }
#pragma unroll
    for (int j = 0; j < 4; ++j) c = wmma_bf16(af[j], bf[j], c);
  }
#pragma unroll
  for (int r = 0; r < 8; ++r) {
    int m = r + half * 8;
    float v = c[r];
    if (do_relu) v = fmaxf(v, 0.f);
    out[((size_t)b * 512 + mt * 16 + m) * 64 + o] = v;
  }
}

// ---------------------------------------------------------------------------
// Kernel 5: he = concat(h[e0], h[e1]) -> fc1(128->64)+ReLU (WMMA) -> fc2 -> sigmoid
// 64 edge-rows per block; wave = 16-row tile. fc1_w staged transposed.
// ---------------------------------------------------------------------------
__global__ __launch_bounds__(128)
void edge_mlp_kernel(const float* __restrict__ hf, const int* __restrict__ eidx,
                     const float* __restrict__ fc1w, const float* __restrict__ fc1b,
                     const float* __restrict__ fc2w, const float* __restrict__ fc2b,
                     float* __restrict__ outp) {
  __shared__ __attribute__((aligned(16))) unsigned short he_l[64 * 128]; // [row][k]
  __shared__ __attribute__((aligned(16))) unsigned short w1t[64 * 128];  // [o][k]
  __shared__ float z_l[64 * 64];
  __shared__ float b1l[64];
  const int tid = threadIdx.x;
  const int m0 = blockIdx.x * 64;

  __builtin_prefetch(eidx + (m0 & 2047) * 2, 0, 1);

  for (int i = tid; i < 64 * 128; i += 128) {
    int o = i >> 7, k = i & 127;
    w1t[i] = f2bfu(fc1w[k * 64 + o]);          // stage transposed
  }
  if (tid < 64) b1l[tid] = fc1b[tid];
  for (int i = tid; i < 64 * 128; i += 128) {
    int r = i >> 7, cc = i & 127;
    int m = m0 + r;
    int b = m >> 11, e = m & 2047;
    int node = (cc < 64) ? eidx[e * 2] : eidx[e * 2 + 1];
    he_l[i] = f2bfu(hf[((size_t)b * 512 + node) * 64 + (cc & 63)]);
  }
  __syncthreads();

  const int wave = tid >> 5, lane = tid & 31;
  const int half = lane >> 4, ln = lane & 15;
  const unsigned short* at = he_l + (wave * 16 + ln) * 128 + half * 8;
  v16bf af[4];
#pragma unroll
  for (int ks = 0; ks < 4; ++ks) af[ks] = frag_ld(at + ks * 32);  // A reused /nt
  for (int nt = 0; nt < 4; ++nt) {
    const unsigned short* btr = w1t + (nt * 16 + ln) * 128 + half * 8;
    v16bf bf[4];
#pragma unroll
    for (int ks = 0; ks < 4; ++ks) bf[ks] = frag_ld(btr + ks * 32);
    v8f c = {};
#pragma unroll
    for (int ks = 0; ks < 4; ++ks) c = wmma_bf16(af[ks], bf[ks], c);
#pragma unroll
    for (int r = 0; r < 8; ++r) {
      int row = wave * 16 + r + half * 8;
      int o = nt * 16 + ln;
      z_l[row * 64 + o] = fmaxf(c[r] + b1l[o], 0.f);
    }
  }
  __syncthreads();
  if (tid < 64) {
    float acc = fc2b[0];
#pragma unroll
    for (int o = 0; o < 64; ++o) acc += z_l[tid * 64 + o] * fc2w[o];
    outp[m0 + tid] = 1.f / (1.f + expf(-acc));
  }
}

// ---------------------------------------------------------------------------
extern "C" void kernel_launch(void* const* d_in, const int* in_sizes, int n_in,
                              void* d_out, int out_size, void* d_ws, size_t ws_size,
                              hipStream_t stream) {
  const float* x    = (const float*)d_in[0];
  const int*   adj  = (const int*)d_in[1];
  const int*   eidx = (const int*)d_in[2];
  const float* w1   = (const float*)d_in[3];
  const float* b1   = (const float*)d_in[4];
  const float* w2   = (const float*)d_in[5];
  const float* b2   = (const float*)d_in[6];
  const float* W1   = (const float*)d_in[7];
  const float* a1   = (const float*)d_in[8];
  const float* W2   = (const float*)d_in[9];
  const float* a2   = (const float*)d_in[10];
  const float* fc1w = (const float*)d_in[11];
  const float* fc1b = (const float*)d_in[12];
  const float* fc2w = (const float*)d_in[13];
  const float* fc2b = (const float*)d_in[14];
  float* outp = (float*)d_out;

  float* ws   = (float*)d_ws;
  float* hbar = ws;            // [4096][64] f32
  float* Wh   = ws + 262144;   // [8][512][64] f32
  float* f1   = ws + 524288;   // [4096]
  float* f2   = ws + 528384;   // [4096]
  float* g1   = ws + 532480;   // [8][512][64] f32
  float* hfin = ws + 794624;   // [8][512][64] f32
  unsigned short* Wht = (unsigned short*)(ws + 1056768); // [8][64][512] bf16

  conv_fused_kernel<<<4096, 128, 0, stream>>>(x, w1, b1, w2, b2, hbar);

  gat_wh_kernel<<<256, 128, 0, stream>>>(hbar, W1, Wh, Wht);
  gat_f_kernel<<<16, 256, 0, stream>>>(Wh, a1, f1, f2);
  gat_attn_kernel<<<256, 128, 0, stream>>>(f1, f2, adj, Wht, g1, 1);

  gat_wh_kernel<<<256, 128, 0, stream>>>(g1, W2, Wh, Wht);
  gat_f_kernel<<<16, 256, 0, stream>>>(Wh, a2, f1, f2);
  gat_attn_kernel<<<256, 128, 0, stream>>>(f1, f2, adj, Wht, hfin, 0);

  edge_mlp_kernel<<<256, 128, 0, stream>>>(hfin, eidx, fc1w, fc1b, fc2w, fc2b, outp);
}
